// SimpleFPCNN_38998303048149
// MI455X (gfx1250) — matrix operand order repaired and checked by reading, not compiled, not run on hardware
//
#include <hip/hip_runtime.h>

typedef _Float16 f16;
typedef _Float16 v16h __attribute__((ext_vector_type(16)));
typedef float    v8f  __attribute__((ext_vector_type(8)));
typedef unsigned int u32x4 __attribute__((ext_vector_type(4)));
typedef int      i32x4 __attribute__((ext_vector_type(4)));
typedef int      i32x8 __attribute__((ext_vector_type(8)));

// Padded f16 activation planes: interior HxW at offset (PAD,PAD), halo = zeros
// (= conv zero padding). PW chosen 8-half aligned.
static const int PAD = 2;
static const int PH  = 256 + 2 * PAD;   // 260
static const int PW  = 264;             // 256 + 2*PAD, padded to 264
static const long long PP = (long long)PH * PW;

// ---------------------------------------------------------------------------
// TDM: load a 3-D f16 tile [tile2][tile1][tile0] from global into LDS.
// D# bitfields per CDNA5 ISA ch.8 (group0: count/lds/global, group1: dims,
// strides in data_size units; groups 2/3 present => up-to-5D form).
// This toolchain exposes the 6-operand builtin:
//   (u32x4 g0, i32x8 g1, i32x4 g2, i32x4 g3, i32x8 extra, i32 cpol)
// ---------------------------------------------------------------------------
__device__ __forceinline__ void tdm_load_3d_f16(
    unsigned ldsOff, const void* gaddr,
    int tile0, int tile1, int tile2,
    long long stride0, long long stride1)
{
    unsigned long long ga = (unsigned long long)(uintptr_t)gaddr;
    u32x4 g0;
    g0[0] = 1u;                                                   // count=1
    g0[1] = ldsOff;                                               // lds_addr
    g0[2] = (unsigned)ga;                                         // global[31:0]
    g0[3] = (unsigned)((ga >> 32) & 0x01FFFFFFull) | (2u << 30);  // [56:32]|type=2

    const unsigned TD0 = 1u << 20, TD1 = 1u << 20, TD2 = 1u << 10;
    unsigned long long s0 = (unsigned long long)stride0;
    unsigned long long s1 = (unsigned long long)stride1;
    i32x8 g1;
    g1[0] = (int)(1u << 16);                                      // data_size=2B
    g1[1] = (int)((TD0 & 0xFFFFu) << 16);                         // tensor_dim0.lo
    g1[2] = (int)(((TD0 >> 16) & 0xFFFFu) | ((TD1 & 0xFFFFu) << 16));
    g1[3] = (int)(((TD1 >> 16) & 0xFFFFu) | (((unsigned)tile0 & 0xFFFFu) << 16));
    g1[4] = (int)(((unsigned)tile1 & 0xFFFFu) | (((unsigned)tile2 & 0xFFFFu) << 16));
    g1[5] = (int)(s0 & 0xFFFFFFFFull);
    g1[6] = (int)(((s0 >> 32) & 0xFFFFull) | ((s1 & 0xFFFFull) << 16));
    g1[7] = (int)((s1 >> 16) & 0xFFFFFFFFull);

    i32x4 g2;
    g2[0] = (int)TD2;   // tensor_dim2
    g2[1] = 0;          // tensor_dim3
    g2[2] = 0;          // tensor_dim2_stride.lo (unused, tile_dim3=0)
    g2[3] = 0;
    i32x4 g3 = {0, 0, 0, 0};
    i32x8 gx = {0, 0, 0, 0, 0, 0, 0, 0};

    __builtin_amdgcn_tensor_load_to_lds(g0, g1, g2, g3, gx, 0);
}

// ---------------------------------------------------------------------------
// Weight packing: OIHW f32 -> per-chunk WMMA-A f16 fragments
// dst[((chunk*16 + m)*32) + kk], k = tap*16 + ci (Cin padded to 16)
// ---------------------------------------------------------------------------
__global__ void __launch_bounds__(256)
pack_weights(const float* __restrict__ w, f16* __restrict__ dst,
             int CoutReal, int CinReal, int ks, int nchunk)
{
    int idx = blockIdx.x * 256 + threadIdx.x;
    if (idx >= nchunk * 512) return;
    int chunk = idx >> 9;
    int rem   = idx & 511;
    int m     = rem >> 5;
    int kk    = rem & 31;
    int T     = ks * ks;
    int tp    = chunk * 2 + (kk >> 4);
    int ci    = kk & 15;
    float v = 0.0f;
    if (tp < T && ci < CinReal && m < CoutReal)
        v = w[(m * CinReal + ci) * T + tp];
    dst[idx] = (f16)v;
}

// ---------------------------------------------------------------------------
// Zero-fill a f16 buffer (re-establishes conv zero-pad halos every call)
// ---------------------------------------------------------------------------
__global__ void __launch_bounds__(256)
fill_zero_f16(f16* __restrict__ p, long long n)
{
    long long i = (long long)blockIdx.x * 256 + threadIdx.x;
    long long stride = (long long)gridDim.x * 256;
    for (; i < n; i += stride) p[i] = (f16)0.0f;
}

// ---------------------------------------------------------------------------
// Implicit-GEMM conv via WMMA.
// Block: 8 waves; wave tile M=16 out-ch x N=16 px; block tile 16x8 px.
// LDS activation tile is channel-planar [ci][ROWS][COLS] (f16).
// INF16: input is padded f16 planes, staged by one TDM descriptor per block.
// else : input is plain f32 (external), staged manually with conversion.
// Output: padded f16 planes (outH) or plain f32 (outF).
// ---------------------------------------------------------------------------
template<int KS, int NCHUNK, bool INF16>
__global__ void __launch_bounds__(256)
conv_wmma(const void* __restrict__ inv, const f16* __restrict__ wpk,
          const float* __restrict__ bias, f16* __restrict__ outH,
          float* __restrict__ outF, int Cin, int CoutReal, int H, int W, int relu)
{
    constexpr int R    = KS / 2;
    constexpr int ROWS = 8 + 2 * R;
    constexpr int COLS = 16 + 2 * R;
    constexpr int RC   = ROWS * COLS;
    constexpr int T    = KS * KS;

    __shared__ f16 sIn[16 * RC];       // [ci][row][col]
    __shared__ f16 sW[NCHUNK * 512];

    const int tid = threadIdx.x;
    const int img = blockIdx.z;
    const int y0  = blockIdx.y * 8  - R;
    const int x0  = blockIdx.x * 16 - R;

    if (tid == 0) __builtin_prefetch(wpk, 0, 0);

    if (INF16) {
        // One TDM 3-D tile per block: [16 ch][ROWS][COLS] from padded planes.
        if (tid < 32) {
            const f16* gp = (const f16*)inv +
                ((size_t)(img * 16) * PH + (size_t)(y0 + PAD)) * PW + (x0 + PAD);
            tdm_load_3d_f16((unsigned)(uintptr_t)&sIn[0], gp,
                            COLS, ROWS, 16, (long long)PW, PP);
            __builtin_amdgcn_s_wait_tensorcnt(0);
        }
    } else {
        const float* in = (const float*)inv;
        for (int idx = tid; idx < 16 * RC; idx += 256) {
            int ci  = idx / RC;
            int rem = idx % RC;
            int row = rem / COLS, col = rem % COLS;
            int gy = y0 + row, gx = x0 + col;
            float v = 0.0f;
            if (ci < Cin && gy >= 0 && gy < H && gx >= 0 && gx < W)
                v = in[((img * Cin + ci) * H + gy) * W + gx];
            sIn[idx] = (f16)v;   // planar: idx = (ci*ROWS+row)*COLS+col
        }
    }
    for (int idx = tid; idx < NCHUNK * 512; idx += 256)
        sW[idx] = wpk[idx];
    __syncthreads();

    const int lane = tid & 31;
    const int wave = tid >> 5;        // output row within block tile
    const int hi   = lane >> 4;       // lane half: selects K 0..15 / 16..31
    const int px   = lane & 15;       // B: pixel column, A: out-channel row

    v8f acc = {};
#pragma unroll
    for (int c0 = 0; c0 < NCHUNK; ++c0) {
        int tpa = 2 * c0;     if (tpa > T - 1) tpa = T - 1;
        int tpb = 2 * c0 + 1; if (tpb > T - 1) tpb = T - 1;
        const int tp = hi ? tpb : tpa;
        const int ky = tp / KS, kx = tp % KS;
        const int rc = (wave + ky) * COLS + (px + kx);

        v16h a, b;
#pragma unroll
        for (int j = 0; j < 8; ++j) {
            b[2 * j]     = sIn[(2 * j) * RC + rc];       // ci = 2j
            b[2 * j + 1] = sIn[(2 * j + 1) * RC + rc];   // ci = 2j+1
            const int kb    = (hi << 4) + 2 * j;
            const int wbase = (c0 * 16 + px) * 32 + kb;  // m = lane & 15
            a[2 * j]     = sW[wbase];
            a[2 * j + 1] = sW[wbase + 1];
        }
        acc = __builtin_amdgcn_wmma_f32_16x16x32_f16(
                  false, a, false, b, (short)0, acc, false, false);
    }

    const int gy = blockIdx.y * 8 + wave;
    const int gx = blockIdx.x * 16 + px;
#pragma unroll
    for (int r = 0; r < 8; ++r) {
        int mch = r + (hi << 3);                  // C/D: VGPR r + lane half
        if (mch < CoutReal) {
            float v = acc[r] + bias[mch];
            if (relu) v = fmaxf(v, 0.0f);
            if (outH)
                outH[((size_t)(img * 16 + mch) * PH + (gy + PAD)) * PW + (gx + PAD)] = (f16)v;
            else
                outF[((size_t)(img * CoutReal + mch) * H + gy) * W + gx] = v;
        }
    }
}

// ---------------------------------------------------------------------------
// Channel softmax over NF=4 (plain f32 planes)
// ---------------------------------------------------------------------------
__global__ void __launch_bounds__(256)
softmax4(float* __restrict__ p, int HW, int nimg)
{
    int i = blockIdx.x * 256 + threadIdx.x;
    if (i >= nimg * HW) return;
    int img = i / HW, pp = i % HW;
    float* b = p + (size_t)img * 4 * HW + pp;
    float v0 = b[0], v1 = b[HW], v2 = b[2 * HW], v3 = b[3 * HW];
    float mx = fmaxf(fmaxf(v0, v1), fmaxf(v2, v3));
    float e0 = __expf(v0 - mx), e1 = __expf(v1 - mx);
    float e2 = __expf(v2 - mx), e3 = __expf(v3 - mx);
    float inv = 1.0f / (e0 + e1 + e2 + e3);
    b[0] = e0 * inv; b[HW] = e1 * inv; b[2 * HW] = e2 * inv; b[3 * HW] = e3 * inv;
}

// ---------------------------------------------------------------------------
// 7x7 bilateral (edge clamp) on padded-f16 h; writes padded-f16 dec_in:
// channels 0..5 = filtered, 6..15 = fixed (center h broadcast).
// ---------------------------------------------------------------------------
__global__ void __launch_bounds__(256)
bilateral_dec_in(const f16* __restrict__ hp, const float* __restrict__ bw,
                 f16* __restrict__ dec_in, int H, int W)
{
    constexpr int RB = 3;
    constexpr int TR = 16 + 2 * RB;          // 22
    __shared__ float sH[TR * TR * 16];
    __shared__ float sBw[64];

    const int tid = threadIdx.x;
    const int img = blockIdx.z;
    const int y0  = blockIdx.y * 16 - RB;
    const int x0  = blockIdx.x * 16 - RB;

    if (tid < 64) sBw[tid] = bw[tid];
    for (int idx = tid; idx < TR * TR * 16; idx += 256) {
        int c   = idx & 15;
        int pos = idx >> 4;
        int col = pos % TR, row = pos / TR;
        int gy = min(max(y0 + row, 0), H - 1);   // edge clamp (interior only)
        int gx = min(max(x0 + col, 0), W - 1);
        sH[idx] = (float)hp[((size_t)(img * 16 + c) * PH + (gy + PAD)) * PW + (gx + PAD)];
    }
    __syncthreads();

    const int py = tid >> 4, px = tid & 15;
    float cen[16];
    const int cbase = ((py + RB) * TR + (px + RB)) * 16;
#pragma unroll
    for (int c = 0; c < 16; ++c) cen[c] = sH[cbase + c];

    float num[4][6] = {};
    float den[4] = {};
    const float inv2s2 = 1.0f / (2.0f * RB * RB);

    for (int dy = 0; dy < 7; ++dy) {
        for (int dx = 0; dx < 7; ++dx) {
            const int sbase = ((py + dy) * TR + (px + dx)) * 16;
            float d2[4] = {};
            float sh6[6];
#pragma unroll
            for (int c = 0; c < 16; ++c) {
                float sv = sH[sbase + c];
                if (c < 6) sh6[c] = sv;
                float dd = cen[c] - sv;
                dd = dd * dd;
#pragma unroll
                for (int f = 0; f < 4; ++f) d2[f] += sBw[f * 16 + c] * dd;
            }
            float sp = (float)((dy - RB) * (dy - RB) + (dx - RB) * (dx - RB)) * inv2s2;
#pragma unroll
            for (int f = 0; f < 4; ++f) {
                float wgt = __expf(-(d2[f] + sp));
                den[f] += wgt;
#pragma unroll
                for (int c = 0; c < 6; ++c) num[f][c] += wgt * sh6[c];
            }
        }
    }

    const int gy = blockIdx.y * 16 + py;
    const int gx = blockIdx.x * 16 + px;
#pragma unroll
    for (int f = 0; f < 4; ++f) {
        float invd = 1.0f / (den[f] + 1e-8f);
        size_t pbase = (size_t)((img * 4 + f) * 16);
#pragma unroll
        for (int c = 0; c < 6; ++c)
            dec_in[((pbase + c) * PH + (gy + PAD)) * PW + (gx + PAD)] = (f16)(num[f][c] * invd);
#pragma unroll
        for (int c = 6; c < 16; ++c)
            dec_in[((pbase + c) * PH + (gy + PAD)) * PW + (gx + PAD)] = (f16)cen[c];
    }
}

// ---------------------------------------------------------------------------
// out = albedo * sum_f pred_f * low_f   (reshaped to (4, 6, H, W))
// ---------------------------------------------------------------------------
__global__ void __launch_bounds__(256)
final_combine(const float* __restrict__ pred, const float* __restrict__ low,
              const float* __restrict__ input, float* __restrict__ out, int HW)
{
    int i = blockIdx.x * 256 + threadIdx.x;
    if (i >= 8 * HW) return;
    int img = i / HW, p = i % HW;
    float pr[4];
#pragma unroll
    for (int f = 0; f < 4; ++f)
        pr[f] = pred[((size_t)img * 4 + f) * HW + p];
    int b = img >> 1, fs = img & 1;
#pragma unroll
    for (int c = 0; c < 3; ++c) {
        float s = 0.0f;
#pragma unroll
        for (int f = 0; f < 4; ++f)
            s += pr[f] * low[((size_t)(img * 4 + f) * 3 + c) * HW + p];
        float alb = input[((size_t)img * 12 + 3 + c) * HW + p];
        out[((size_t)(b * 6 + fs * 3 + c)) * HW + p] = alb * s;
    }
}

// ---------------------------------------------------------------------------
// Orchestration
// ---------------------------------------------------------------------------
extern "C" void kernel_launch(void* const* d_in, const int* in_sizes, int n_in,
                              void* d_out, int out_size, void* d_ws, size_t ws_size,
                              hipStream_t stream)
{
    const float* input  = (const float*)d_in[0];
    const float* enc1_w = (const float*)d_in[1];
    const float* enc1_b = (const float*)d_in[2];
    const float* enc2_w = (const float*)d_in[3];
    const float* enc2_b = (const float*)d_in[4];
    const float* dec1_w = (const float*)d_in[5];
    const float* dec1_b = (const float*)d_in[6];
    const float* dec2_w = (const float*)d_in[7];
    const float* dec2_b = (const float*)d_in[8];
    const float* fp1_w  = (const float*)d_in[9];
    const float* fp1_b  = (const float*)d_in[10];
    const float* fp2_w  = (const float*)d_in[11];
    const float* fp2_b  = (const float*)d_in[12];
    const float* bw     = (const float*)d_in[13];

    const int H = 256, W = 256, HW = H * W;

    char* ws = (char*)d_ws;
    size_t off = 0;
    auto alloc = [&](size_t bytes) -> void* {
        void* p = ws + off;
        off = (off + bytes + 255) & ~(size_t)255;
        return p;
    };

    f16* wpk_enc1 = (f16*)alloc((size_t)5  * 512 * sizeof(f16));
    f16* wpk_enc2 = (f16*)alloc((size_t)5  * 512 * sizeof(f16));
    f16* wpk_fp1  = (f16*)alloc((size_t)13 * 512 * sizeof(f16));
    f16* wpk_fp2  = (f16*)alloc((size_t)13 * 512 * sizeof(f16));
    f16* wpk_dec1 = (f16*)alloc((size_t)5  * 512 * sizeof(f16));
    f16* wpk_dec2 = (f16*)alloc((size_t)5  * 512 * sizeof(f16));

    const long long nH = (long long)8  * 16 * PP;   // padded f16 planes
    const long long nT = (long long)32 * 16 * PP;
    const long long nD = (long long)32 * 16 * PP;
    f16*   hbuf16 = (f16*)alloc((size_t)nH * sizeof(f16));
    f16*   tbuf16 = (f16*)alloc((size_t)nT * sizeof(f16));
    f16*   decin  = (f16*)alloc((size_t)nD * sizeof(f16));
    float* pred   = (float*)alloc((size_t)8  * 4 * HW * sizeof(float));
    float* lowb   = (float*)alloc((size_t)32 * 3 * HW * sizeof(float));

    // 0) re-zero padded buffers (halos = conv zero padding)
    fill_zero_f16<<<4096, 256, 0, stream>>>(hbuf16, nH);
    fill_zero_f16<<<4096, 256, 0, stream>>>(tbuf16, nT);
    fill_zero_f16<<<4096, 256, 0, stream>>>(decin,  nD);

    // 1) pack conv weights into WMMA fragments
    pack_weights<<<(5  * 512 + 255) / 256, 256, 0, stream>>>(enc1_w, wpk_enc1, 16, 12, 3, 5);
    pack_weights<<<(5  * 512 + 255) / 256, 256, 0, stream>>>(enc2_w, wpk_enc2, 16, 16, 3, 5);
    pack_weights<<<(13 * 512 + 255) / 256, 256, 0, stream>>>(fp1_w,  wpk_fp1,  16, 16, 5, 13);
    pack_weights<<<(13 * 512 + 255) / 256, 256, 0, stream>>>(fp2_w,  wpk_fp2,   4, 16, 5, 13);
    pack_weights<<<(5  * 512 + 255) / 256, 256, 0, stream>>>(dec1_w, wpk_dec1, 16, 16, 3, 5);
    pack_weights<<<(5  * 512 + 255) / 256, 256, 0, stream>>>(dec2_w, wpk_dec2,  3, 16, 3, 5);

    // 2) encoder: h = enc2(relu(enc1(x)))
    dim3 cg8(W / 16, H / 8, 8);
    conv_wmma<3, 5, false><<<cg8, 256, 0, stream>>>(input,  wpk_enc1, enc1_b, tbuf16, nullptr, 12, 16, H, W, 1);
    conv_wmma<3, 5, true ><<<cg8, 256, 0, stream>>>(tbuf16, wpk_enc2, enc2_b, hbuf16, nullptr, 16, 16, H, W, 0);

    // 3) pred = softmax(fp2(relu(fp1(h))))
    conv_wmma<5, 13, true><<<cg8, 256, 0, stream>>>(hbuf16, wpk_fp1, fp1_b, tbuf16, nullptr, 16, 16, H, W, 1);
    conv_wmma<5, 13, true><<<cg8, 256, 0, stream>>>(tbuf16, wpk_fp2, fp2_b, nullptr, pred,   16,  4, H, W, 0);
    softmax4<<<(8 * HW + 255) / 256, 256, 0, stream>>>(pred, HW, 8);

    // 4) bilateral -> dec_in (filtered 0..5 + fixed 6..15)
    bilateral_dec_in<<<dim3(W / 16, H / 16, 8), 256, 0, stream>>>(hbuf16, bw, decin, H, W);

    // 5) decoder on 32 virtual images
    dim3 cg32(W / 16, H / 8, 32);
    conv_wmma<3, 5, true><<<cg32, 256, 0, stream>>>(decin,  wpk_dec1, dec1_b, tbuf16, nullptr, 16, 16, H, W, 1);
    conv_wmma<3, 5, true><<<cg32, 256, 0, stream>>>(tbuf16, wpk_dec2, dec2_b, nullptr, lowb,   16,  3, H, W, 0);

    // 6) final blend
    final_combine<<<(8 * HW + 255) / 256, 256, 0, stream>>>(pred, lowb, input, (float*)d_out, HW);
}
